// MultiHeadedSelfAttention_26233660244012
// MI455X (gfx1250) — compile-verified, matrix operand order checked by reading, not run on hardware
//
#include <hip/hip_runtime.h>
#include <hip/hip_bf16.h>

// ---------------------------------------------------------------------------
// MultiHeadedSelfAttention for MI455X (gfx1250, wave32, WMMA bf16 w/ f32 acc)
// B=64, S=512, E=1024, H=16, D=64
// - All WMMA operands stored K-major => every 16x32 fragment = two b128 loads
// - Double-buffered GEMM tiles filled with GLOBAL_LOAD_ASYNC_TO_LDS_B128
//   (ASYNCcnt) so the fill overlaps the WMMA stream.
// ---------------------------------------------------------------------------

#define BATCH 64
#define SEQ   512
#define EMB   1024
#define HEADS 16
#define HDIM  64

typedef __attribute__((ext_vector_type(16))) __bf16 bf16x16;
typedef __attribute__((ext_vector_type(8)))  float  f32x8;

union FragU {           // two 128-bit loads -> one WMMA A/B fragment, no packing
  uint4   q[2];
  bf16x16 f;
};

static __device__ __forceinline__ unsigned short f32_to_bf16_bits(float f) {
  unsigned int u = __float_as_uint(f);
  u += 0x7fffu + ((u >> 16) & 1u);            // round-to-nearest-even
  return (unsigned short)(u >> 16);
}

// CDNA5 async DMA: global -> LDS, 128-bit per lane, tracked by ASYNCcnt.
static __device__ __forceinline__ void async_load_b128(void* lds_dst,
                                                       const void* gsrc) {
  unsigned lds_off = (unsigned)(size_t)lds_dst;            // addr[31:0] = LDS offset
  unsigned long long ga = (unsigned long long)(size_t)gsrc;
  asm volatile("global_load_async_to_lds_b128 %0, %1, off"
               :: "v"(lds_off), "v"(ga)
               : "memory");
}
static __device__ __forceinline__ void wait_async0() {
  asm volatile("s_wait_asynccnt 0x0" ::: "memory");
}

// ---------------------------------------------------------------------------
// Phase 0a: elementwise fp32 -> bf16 (activations, row-major kept)
// ---------------------------------------------------------------------------
__global__ void cast_f32_to_bf16(const float* __restrict__ in,
                                 unsigned short* __restrict__ out, int n) {
  int i = blockIdx.x * 256 + threadIdx.x;
  if (i < n) out[i] = f32_to_bf16_bits(in[i]);
}

// ---------------------------------------------------------------------------
// Phase 0b: fp32 [E,E] -> bf16 transposed [E,E] (weights become K-major)
// ---------------------------------------------------------------------------
__global__ __launch_bounds__(256) void cast_transpose_bf16(
    const float* __restrict__ in, unsigned short* __restrict__ out) {
  __shared__ unsigned short t[32][33];
  const int tx = threadIdx.x & 31;
  const int ty = threadIdx.x >> 5;            // 0..7
  const int bx = blockIdx.x * 32;
  const int by = blockIdx.y * 32;
#pragma unroll
  for (int i = 0; i < 4; ++i) {
    int r = by + ty + i * 8;
    t[ty + i * 8][tx] = f32_to_bf16_bits(in[(size_t)r * EMB + bx + tx]);
  }
  __syncthreads();
#pragma unroll
  for (int i = 0; i < 4; ++i) {
    int r = bx + ty + i * 8;                  // out row = in col
    out[(size_t)r * EMB + by + tx] = t[tx][ty + i * 8];
  }
}

// ---------------------------------------------------------------------------
// Phase 1/3: bf16 GEMM  C[M,N] = A[M,K] * Bt[N,K]^T   (Bt is K-major)
//   mode 0: bf16 out, [B,H,S,D] (Q,K) | mode 2: bf16 out, [B,H,D,S] (V)
//   mode 1: fp32 out, row-major + bias (output projection)
// 256 thr = 8 waves; block tile 128x128, wave tile 32x64, K-step 32.
// Double-buffered LDS (K-major, stride 20 dw) filled by async b128 DMA.
// 8 WMMA per wave per K-step.
// ---------------------------------------------------------------------------
__global__ __launch_bounds__(256) void gemm_bf16(
    const unsigned short* __restrict__ A,   // [M,K] row-major
    const unsigned short* __restrict__ Bt,  // [N,K] row-major (transposed W)
    void* __restrict__ Cout,
    const float* __restrict__ bias,
    int M, int N, int K, int mode) {
  __shared__ uint4 As4[2][128 * 5];   // ping-pong, 128 rows x (16+4 pad) dwords
  __shared__ uint4 Bs4[2][128 * 5];

  const int tid  = threadIdx.x;
  const int lane = tid & 31;
  const int wave = tid >> 5;
  const int wm   = wave & 3;      // 4 waves along M, 32 rows each
  const int wn   = wave >> 2;     // 2 waves along N, 64 cols each
  const int half = lane >> 4;
  const int l16  = lane & 15;

  const int mBase = blockIdx.x * 128;
  const int nBase = blockIdx.y * 128;

  f32x8 acc[2][4];
#pragma unroll
  for (int s = 0; s < 2; ++s)
#pragma unroll
    for (int t = 0; t < 4; ++t) acc[s][t] = (f32x8){};

  // cooperative fill: 512 uint4 per tile -> 2 per thread per matrix
  const int r0 = tid >> 1;            // row 0..127
  const int c0 = (tid & 1) * 2;       // uint4 col {0,1} or {2,3}

  // preload K-step 0 into buffer 0
#pragma unroll
  for (int j = 0; j < 2; ++j) {
    const int c4 = c0 + j;
    async_load_b128(&As4[0][r0 * 5 + c4], A  + (size_t)(mBase + r0) * K + c4 * 8);
    async_load_b128(&Bs4[0][r0 * 5 + c4], Bt + (size_t)(nBase + r0) * K + c4 * 8);
  }
  wait_async0();
  __syncthreads();

  int cur = 0;
  for (int k0 = 0; k0 < K; k0 += 32) {
    const int  nxt  = cur ^ 1;
    const bool more = (k0 + 32) < K;
    if (more) {                       // fill next buffer while computing
#pragma unroll
      for (int j = 0; j < 2; ++j) {
        const int c4 = c0 + j;
        async_load_b128(&As4[nxt][r0 * 5 + c4],
                        A  + (size_t)(mBase + r0) * K + (k0 + 32) + c4 * 8);
        async_load_b128(&Bs4[nxt][r0 * 5 + c4],
                        Bt + (size_t)(nBase + r0) * K + (k0 + 32) + c4 * 8);
      }
    }

    FragU af[2];
#pragma unroll
    for (int s = 0; s < 2; ++s) {
      const int arow = wm * 32 + s * 16 + l16;
      af[s].q[0] = As4[cur][arow * 5 + half];
      af[s].q[1] = As4[cur][arow * 5 + 2 + half];
    }
#pragma unroll
    for (int t = 0; t < 4; ++t) {
      FragU bf;
      const int n = wn * 64 + t * 16 + l16;
      bf.q[0] = Bs4[cur][n * 5 + half];
      bf.q[1] = Bs4[cur][n * 5 + 2 + half];
      acc[0][t] = __builtin_amdgcn_wmma_f32_16x16x32_bf16(false, af[0].f, false, bf.f,
                                                          (short)0, acc[0][t], false, false);
      acc[1][t] = __builtin_amdgcn_wmma_f32_16x16x32_bf16(false, af[1].f, false, bf.f,
                                                          (short)0, acc[1][t], false, false);
    }

    if (more) wait_async0();
    __syncthreads();
    cur = nxt;
  }

  // epilogue: C layout VGPR v <-> row v + 8*half, col l16
#pragma unroll
  for (int s = 0; s < 2; ++s) {
#pragma unroll
    for (int v = 0; v < 8; ++v) {
      const int row = mBase + wm * 32 + s * 16 + v + 8 * half;
#pragma unroll
      for (int t = 0; t < 4; ++t) {
        const int col = nBase + wn * 64 + t * 16 + l16;
        const float val = acc[s][t][v];
        if (mode == 1) {
          ((float*)Cout)[(size_t)row * N + col] = val + bias[col];
        } else {
          int b = row >> 9, ss = row & 511;
          int h = col >> 6, d = col & 63;
          size_t idx;
          if (mode == 0)        // [B,H,S,D]
            idx = (((size_t)(b * HEADS + h) * SEQ) + ss) * HDIM + d;
          else                  // mode 2: [B,H,D,S]
            idx = (((size_t)(b * HEADS + h) * HDIM) + d) * SEQ + ss;
          ((unsigned short*)Cout)[idx] = f32_to_bf16_bits(val);
        }
      }
    }
  }
}

// ---------------------------------------------------------------------------
// Phase 2: causal flash attention, one block per (b,h).
// K [512,64] and V^T [64,512] staged ONCE in LDS via async b128 DMA.
// 8 waves; each wave owns 16-row query tiles; online softmax via wave32
// shfl_xor over 16-lane C-layout row groups; every fragment = two b128.
// ---------------------------------------------------------------------------
__global__ __launch_bounds__(256) void attn_causal(
    const unsigned short* __restrict__ Q,    // [B,H,S,D]
    const unsigned short* __restrict__ Km,   // [B,H,S,D]
    const unsigned short* __restrict__ Vt,   // [B,H,D,S]
    unsigned short* __restrict__ Ctx) {      // [B,S,H,D]
  __shared__ uint4 Ks4[SEQ * 9];      // key rows: 8 data + 1 pad uint4 = 72KB
  __shared__ uint4 Vs4[HDIM * 65];    // d rows: 64 data + 1 pad uint4 = 65KB
  __shared__ uint4 Ps4[8][16 * 5];    // per-wave 16x32 bf16, stride 20 dw

  const int tid  = threadIdx.x;
  const int lane = tid & 31;
  const int wave = tid >> 5;
  const int half = lane >> 4;
  const int l16  = lane & 15;

  const int bh = blockIdx.x;
  const int b  = bh >> 4;
  const int h  = bh & 15;
  const size_t base = (size_t)bh * SEQ * HDIM;

  // async DMA staging of K and V^T
  {
    const uint4* kg = (const uint4*)(Km + base);
    const uint4* vg = (const uint4*)(Vt + base);
    for (int i = tid; i < SEQ * 8; i += 256) {        // 4096 uint4
      int key = i >> 3, c = i & 7;
      async_load_b128(&Ks4[key * 9 + c], kg + i);
    }
    for (int i = tid; i < HDIM * 64; i += 256) {      // 4096 uint4
      int d = i >> 6, c = i & 63;
      async_load_b128(&Vs4[d * 65 + c], vg + i);
    }
    wait_async0();
  }
  __syncthreads();

  const uint4* qg4 = (const uint4*)(Q + base);        // 8 uint4 per row

  for (int qt = wave; qt < SEQ / 16; qt += 8) {       // wave-uniform
    const int qbase = qt * 16;

    FragU qa0, qa1;                                   // Q 16x64 = two fragments
    {
      const int qrow = qbase + l16;
      qa0.q[0] = qg4[qrow * 8 + half];
      qa0.q[1] = qg4[qrow * 8 + 2 + half];
      qa1.q[0] = qg4[qrow * 8 + 4 + half];
      qa1.q[1] = qg4[qrow * 8 + 6 + half];
    }

    float rowm[8], rows[8];
#pragma unroll
    for (int v = 0; v < 8; ++v) { rowm[v] = -1e30f; rows[v] = 0.0f; }
    f32x8 o0 = {}, o1 = {}, o2 = {}, o3 = {};

    const int kbMax = (qbase + 15) >> 5;              // causal, wave-uniform
    for (int kb = 0; kb <= kbMax; ++kb) {
      // ---- scores: S[16q x 32k] over d=64 ----
      f32x8 s0 = {}, s1 = {};
#pragma unroll
      for (int sub = 0; sub < 2; ++sub) {
        const int key = kb * 32 + sub * 16 + l16;
        FragU kf0, kf1;                               // d 0..31 / 32..63
        kf0.q[0] = Ks4[key * 9 + half];
        kf0.q[1] = Ks4[key * 9 + 2 + half];
        kf1.q[0] = Ks4[key * 9 + 4 + half];
        kf1.q[1] = Ks4[key * 9 + 6 + half];
        if (sub == 0) {
          s0 = __builtin_amdgcn_wmma_f32_16x16x32_bf16(false, qa0.f, false, kf0.f,
                                                       (short)0, s0, false, false);
          s0 = __builtin_amdgcn_wmma_f32_16x16x32_bf16(false, qa1.f, false, kf1.f,
                                                       (short)0, s0, false, false);
        } else {
          s1 = __builtin_amdgcn_wmma_f32_16x16x32_bf16(false, qa0.f, false, kf0.f,
                                                       (short)0, s1, false, false);
          s1 = __builtin_amdgcn_wmma_f32_16x16x32_bf16(false, qa1.f, false, kf1.f,
                                                       (short)0, s1, false, false);
        }
      }

      // ---- causal mask + online softmax ----
      unsigned short* pw = (unsigned short*)Ps4[wave];
#pragma unroll
      for (int v = 0; v < 8; ++v) {
        const int qrow = qbase + v + 8 * half;
        const int key0 = kb * 32 + l16;
        float x0 = (key0 <= qrow) ? s0[v] : -1e30f;
        float x1 = (key0 + 16 <= qrow) ? s1[v] : -1e30f;
        float mx = fmaxf(x0, x1);
        mx = fmaxf(mx, __shfl_xor(mx, 1, 32));
        mx = fmaxf(mx, __shfl_xor(mx, 2, 32));
        mx = fmaxf(mx, __shfl_xor(mx, 4, 32));
        mx = fmaxf(mx, __shfl_xor(mx, 8, 32));
        float mnew  = fmaxf(rowm[v], mx);
        float scale = __expf(rowm[v] - mnew);
        float p0 = __expf(x0 - mnew);
        float p1 = __expf(x1 - mnew);
        float sl = p0 + p1;
        sl += __shfl_xor(sl, 1, 32);
        sl += __shfl_xor(sl, 2, 32);
        sl += __shfl_xor(sl, 4, 32);
        sl += __shfl_xor(sl, 8, 32);
        rows[v] = rows[v] * scale + sl;
        rowm[v] = mnew;
        o0[v] *= scale; o1[v] *= scale; o2[v] *= scale; o3[v] *= scale;
        pw[(v + 8 * half) * 40 + l16]      = f32_to_bf16_bits(p0);
        pw[(v + 8 * half) * 40 + 16 + l16] = f32_to_bf16_bits(p1);
      }

      // ---- reload P as A-fragment (wave-local LDS, in-order) ----
      FragU pa;
      pa.q[0] = Ps4[wave][l16 * 5 + half];
      pa.q[1] = Ps4[wave][l16 * 5 + 2 + half];

      // ---- PV: out[16q x 64d] += P[16x32] * V[32k x 64d] from V^T LDS ----
#pragma unroll
      for (int c = 0; c < 4; ++c) {
        const int dcol = c * 16 + l16;
        FragU vf;
        vf.q[0] = Vs4[dcol * 65 + kb * 4 + half];
        vf.q[1] = Vs4[dcol * 65 + kb * 4 + 2 + half];
        if (c == 0)      o0 = __builtin_amdgcn_wmma_f32_16x16x32_bf16(false, pa.f, false, vf.f, (short)0, o0, false, false);
        else if (c == 1) o1 = __builtin_amdgcn_wmma_f32_16x16x32_bf16(false, pa.f, false, vf.f, (short)0, o1, false, false);
        else if (c == 2) o2 = __builtin_amdgcn_wmma_f32_16x16x32_bf16(false, pa.f, false, vf.f, (short)0, o2, false, false);
        else             o3 = __builtin_amdgcn_wmma_f32_16x16x32_bf16(false, pa.f, false, vf.f, (short)0, o3, false, false);
      }
    }

    // ---- normalize + store context bf16 as [B,S,H,D] (row-major [BS,E]) ----
#pragma unroll
    for (int v = 0; v < 8; ++v) {
      const int s = qbase + v + 8 * half;
      const float inv = 1.0f / rows[v];
      const size_t rbase = (((size_t)(b * SEQ + s)) * HEADS + h) * HDIM;
      Ctx[rbase + 0 * 16 + l16] = f32_to_bf16_bits(o0[v] * inv);
      Ctx[rbase + 1 * 16 + l16] = f32_to_bf16_bits(o1[v] * inv);
      Ctx[rbase + 2 * 16 + l16] = f32_to_bf16_bits(o2[v] * inv);
      Ctx[rbase + 3 * 16 + l16] = f32_to_bf16_bits(o3[v] * inv);
    }
  }
}

// ---------------------------------------------------------------------------
// Launch: cast(+transpose W) -> 3x GEMM (Q,K,V^T) -> attention -> GEMM+bias
// Workspace (bf16 elems): X/ctx(reused) 32M | 4 W^T 4M | Q,K,V 3x32M  ~264MB
// ---------------------------------------------------------------------------
extern "C" void kernel_launch(void* const* d_in, const int* in_sizes, int n_in,
                              void* d_out, int out_size, void* d_ws, size_t ws_size,
                              hipStream_t stream) {
  const float* X  = (const float*)d_in[0];
  const float* Wq = (const float*)d_in[1];
  const float* Wk = (const float*)d_in[2];
  const float* Wv = (const float*)d_in[3];
  const float* Wo = (const float*)d_in[4];
  const float* bo = (const float*)d_in[5];

  const int M  = BATCH * SEQ;     // 32768
  const int E2 = EMB * EMB;       // 1048576
  const int XN = M * EMB;         // 33554432

  unsigned short* ws   = (unsigned short*)d_ws;
  unsigned short* Xbf  = ws;                      // 32M (reused as ctx)
  unsigned short* Wqt  = ws + (size_t)XN;
  unsigned short* Wkt  = Wqt + (size_t)E2;
  unsigned short* Wvt  = Wkt + (size_t)E2;
  unsigned short* Wot  = Wvt + (size_t)E2;
  unsigned short* Qbf  = Wot + (size_t)E2;
  unsigned short* Kbf  = Qbf + (size_t)XN;
  unsigned short* Vbf  = Kbf + (size_t)XN;
  unsigned short* Ctx  = Xbf;                     // X dead after QKV GEMMs

  // phase 0: casts (weights transposed to K-major)
  cast_f32_to_bf16<<<(XN + 255) / 256, 256, 0, stream>>>(X, Xbf, XN);
  dim3 tgrid(EMB / 32, EMB / 32, 1);
  cast_transpose_bf16<<<tgrid, 256, 0, stream>>>(Wq, Wqt);
  cast_transpose_bf16<<<tgrid, 256, 0, stream>>>(Wk, Wkt);
  cast_transpose_bf16<<<tgrid, 256, 0, stream>>>(Wv, Wvt);
  cast_transpose_bf16<<<tgrid, 256, 0, stream>>>(Wo, Wot);

  // phase 1: projections  (Q,K -> [B,H,S,D];  V -> [B,H,D,S])
  dim3 ggrid(M / 128, EMB / 128, 1);
  gemm_bf16<<<ggrid, 256, 0, stream>>>(Xbf, Wqt, (void*)Qbf, nullptr, M, EMB, EMB, 0);
  gemm_bf16<<<ggrid, 256, 0, stream>>>(Xbf, Wkt, (void*)Kbf, nullptr, M, EMB, EMB, 0);
  gemm_bf16<<<ggrid, 256, 0, stream>>>(Xbf, Wvt, (void*)Vbf, nullptr, M, EMB, EMB, 2);

  // phase 2: causal attention (ctx overwrites X region)
  attn_causal<<<BATCH * HEADS, 256, 0, stream>>>(Qbf, Kbf, Vbf, Ctx);

  // phase 3: output projection + bias -> fp32 d_out
  gemm_bf16<<<ggrid, 256, 0, stream>>>(Ctx, Wot, d_out, bo, M, EMB, EMB, 1);
}
